// QLSTM_65481071402632
// MI455X (gfx1250) — matrix operand order repaired and checked by reading, not compiled, not run on hardware
//
#include <hip/hip_runtime.h>
#include <hip/hip_bf16.h>

// Problem dims (fixed by reference)
#define T_ 256
#define B_ 256
#define I_ 256
#define H_ 512

typedef __attribute__((ext_vector_type(16))) __bf16 v16bf;
typedef __attribute__((ext_vector_type(8)))  float  v8f;

union Frag32 { uint4 u[2]; v16bf v; };

// Scheduling-control: pin the software pipeline so RA can't collapse it.
// masks: 0x020 = VMEM read, 0x100 = DS read, 0x008 = MFMA/WMMA
#if defined(__has_builtin)
#if __has_builtin(__builtin_amdgcn_sched_group_barrier)
#define SGB(mask, size, id) __builtin_amdgcn_sched_group_barrier((mask), (size), (id))
#endif
#endif
#ifndef SGB
#define SGB(mask, size, id)
#endif

static __device__ __forceinline__ unsigned short f32_to_bf16(float f) {
    unsigned int u = __float_as_uint(f);
    u += 0x7FFFu + ((u >> 16) & 1u);       // round-to-nearest-even
    return (unsigned short)(u >> 16);
}

static __device__ __forceinline__ float sigmoidf_(float x) {
    return 1.0f / (1.0f + __expf(-x));
}

// ---------------------------------------------------------------------------
// Prep 1: fuse the four recurrent weights into one bf16 table
//         wcat[g][n][k] = bf16(W{g}_r[n][k]),  g in {f,i,g,o}, shape [4*H][H]
// ---------------------------------------------------------------------------
__global__ void qlstm_prep_w(const float* __restrict__ Wf,
                             const float* __restrict__ Wi,
                             const float* __restrict__ Wg,
                             const float* __restrict__ Wo,
                             unsigned short* __restrict__ wcat) {
    int idx = blockIdx.x * blockDim.x + threadIdx.x;   // 0 .. 4*H*H-1
    if (idx >= 4 * H_ * H_) return;
    int g   = idx >> 18;                               // H*H = 2^18
    int rem = idx & (H_ * H_ - 1);
    const float* src = (g == 0) ? Wf : (g == 1) ? Wi : (g == 2) ? Wg : Wo;
    wcat[idx] = f32_to_bf16(src[rem]);
}

// ---------------------------------------------------------------------------
// Prep 2: x-part of the quantum angle for all timesteps (parallel over T*B):
//         xdot[t][b][g] = b{g}_l[0] + dot(x[t,b,:], W{g}_l[0, 0:I])
// ---------------------------------------------------------------------------
__global__ void qlstm_prep_xdot(const float* __restrict__ x,
                                const float* __restrict__ wlf, const float* __restrict__ wli,
                                const float* __restrict__ wlg, const float* __restrict__ wlo,
                                const float* __restrict__ blf, const float* __restrict__ bli,
                                const float* __restrict__ blg, const float* __restrict__ blo,
                                float* __restrict__ xdot) {
    int idx = blockIdx.x * blockDim.x + threadIdx.x;   // t*B + b
    if (idx >= T_ * B_) return;
    const float* xr = x + (size_t)idx * I_;
    float s0 = blf[0], s1 = bli[0], s2 = blg[0], s3 = blo[0];
    for (int i = 0; i < I_; ++i) {
        float v = xr[i];
        s0 += v * wlf[i]; s1 += v * wli[i]; s2 += v * wlg[i]; s3 += v * wlo[i];
    }
    float4* o = (float4*)(xdot + (size_t)idx * 4);
    *o = make_float4(s0, s1, s2, s3);
}

// ---------------------------------------------------------------------------
// One LSTM step. Grid (16,4) x 256 threads.
//   blockIdx.x : batch row tile (16 rows)
//   blockIdx.y : 128-wide hidden column chunk; wave w owns cols nb..nb+15
// Each wave computes all four gate GEMM tiles (same A, four B) with
// v_wmma_f32_16x16x32_bf16 in a depth-2 software pipeline, then the fused
// elementwise LSTM update. h ping-pongs through ys; c ping-pongs in ws.
// ---------------------------------------------------------------------------
__global__ __launch_bounds__(256) void qlstm_step(
    int t,
    const unsigned short* __restrict__ wcat,
    const float* __restrict__ xdot,
    const float* __restrict__ wlf, const float* __restrict__ wli,
    const float* __restrict__ wlg, const float* __restrict__ wlo,
    const float* __restrict__ wpf, const float* __restrict__ bpf,
    const float* __restrict__ wpi, const float* __restrict__ bpi,
    const float* __restrict__ wpg, const float* __restrict__ bpg,
    const float* __restrict__ wpo, const float* __restrict__ bpo,
    const float* __restrict__ cprev, float* __restrict__ cnew,
    float* __restrict__ ys, float* __restrict__ out_hx, float* __restrict__ out_cx)
{
    // 512 + 8 bf16 pad per row -> row stride 1040 B (16B aligned, 4-bank step)
    __shared__ __align__(16) unsigned short sA[16][520];
    __shared__ float sQp[16][16][4];
    __shared__ float sQ[4][16];

    const int tid = threadIdx.x;
    const int rb  = blockIdx.x * 16;

    // ---- Phase 1: stage h(t-1) rows to LDS as bf16 + partial q dots -------
    {
        int sm = tid >> 4;            // row 0..15
        int ss = tid & 15;            // 32-wide k slice
        const float* hrow = (t > 0)
            ? ys + ((size_t)(t - 1) * B_ + (rb + sm)) * H_ : ys;
        float q0 = 0.f, q1 = 0.f, q2 = 0.f, q3 = 0.f;
        int kbase = ss * 32;
        for (int kk = 0; kk < 32; ++kk) {
            int k   = kbase + kk;
            float v = (t > 0) ? hrow[k] : 0.0f;
            sA[sm][k] = f32_to_bf16(v);
            q0 += v * wlf[I_ + k];
            q1 += v * wli[I_ + k];
            q2 += v * wlg[I_ + k];
            q3 += v * wlo[I_ + k];
        }
        sQp[sm][ss][0] = q0; sQp[sm][ss][1] = q1;
        sQp[sm][ss][2] = q2; sQp[sm][ss][3] = q3;
    }
    __syncthreads();
    if (tid < 64) {                   // reduce 16 partials -> q[g][m]
        int g = tid & 3, m = tid >> 2;
        float s = 0.f;
        for (int j = 0; j < 16; ++j) s += sQp[m][j][g];
        float xd = xdot[((size_t)t * B_ + (rb + m)) * 4 + g];
        sQ[g][m] = __cosf(xd + s);    // <Z_0> = cos(theta_0)
    }
    __syncthreads();

    // ---- Phase 2: four fused 16x16 WMMA tiles over K = 512 -----------------
    const int lane = tid & 31;
    const int wave = tid >> 5;
    const int lh   = lane >> 4;       // half-wave selector
    const int nb   = blockIdx.y * 128 + wave * 16;
    const int gn   = nb + (lane & 15);

    v8f accF = {}; v8f accI = {}; v8f accG = {}; v8f accO = {};

    if (t > 0) {
        // B layout: lane holds 16 contiguous K values of column gn (khalf=16*lh)
        const unsigned short* bB[4];
        bB[0] = wcat + ((size_t)(0 * H_ + gn)) * H_ + lh * 16;
        bB[1] = wcat + ((size_t)(1 * H_ + gn)) * H_ + lh * 16;
        bB[2] = wcat + ((size_t)(2 * H_ + gn)) * H_ + lh * 16;
        bB[3] = wcat + ((size_t)(3 * H_ + gn)) * H_ + lh * 16;
        // A layout: lane m = lane&15; 16 contiguous K values (khalf=16*lh)
        const unsigned short* aRow = &sA[lane & 15][0];

        // Software pipeline: B prefetched 2 k-blocks ahead (3 rotating
        // register slots x 4 gates), A (LDS, fast) 1 ahead. Fully unrolled;
        // sched_group_barriers pin the issue order (8 VMEM loads for i+2,
        // 2 DS loads for i+1, 4 WMMAs for i) so the register allocator must
        // keep the prefetched fragments live instead of collapsing the
        // pipeline into load->wait(0)->wmma chains.
        Frag32 fa[2];
        Frag32 fB[3][4];

        auto loadA = [&](int i, int slot) {
            int kb = i * 32;
            fa[slot].u[0] = *(const uint4*)(aRow + kb + 16 * lh);
            fa[slot].u[1] = *(const uint4*)(aRow + kb + 16 * lh + 8);
        };
        auto loadB = [&](int i, int slot) {
            int kb = i * 32;
#pragma unroll
            for (int g = 0; g < 4; ++g) {
                fB[slot][g].u[0] = *(const uint4*)(bB[g] + kb);
                fB[slot][g].u[1] = *(const uint4*)(bB[g] + kb + 8);
            }
        };

        loadB(0, 0);
        SGB(0x020, 8, 0);
        loadB(1, 1);
        SGB(0x020, 8, 0);
        loadA(0, 0);
        SGB(0x100, 2, 0);

#pragma unroll
        for (int i = 0; i < 16; ++i) {
            const int cs = i % 3;             // current B slot
            const int as = i & 1;             // current A slot
            if (i + 2 < 16) { loadB(i + 2, (i + 2) % 3); SGB(0x020, 8, 0); }
            if (i + 1 < 16) { loadA(i + 1, (i + 1) & 1); SGB(0x100, 2, 0); }
            accF = __builtin_amdgcn_wmma_f32_16x16x32_bf16(false, fa[as].v, false,
                       fB[cs][0].v, (short)0, accF, false, false);
            accI = __builtin_amdgcn_wmma_f32_16x16x32_bf16(false, fa[as].v, false,
                       fB[cs][1].v, (short)0, accI, false, false);
            accG = __builtin_amdgcn_wmma_f32_16x16x32_bf16(false, fa[as].v, false,
                       fB[cs][2].v, (short)0, accG, false, false);
            accO = __builtin_amdgcn_wmma_f32_16x16x32_bf16(false, fa[as].v, false,
                       fB[cs][3].v, (short)0, accO, false, false);
            SGB(0x008, 4, 0);
        }
    }

    // ---- Phase 3: fused gate nonlinearity + cell/hidden update -------------
    float vwpf = wpf[gn], vbpf = bpf[gn];
    float vwpi = wpi[gn], vbpi = bpi[gn];
    float vwpg = wpg[gn], vbpg = bpg[gn];
    float vwpo = wpo[gn], vbpo = bpo[gn];

    float* ysrow = ys + (size_t)t * B_ * H_;

#pragma unroll
    for (int r = 0; r < 8; ++r) {
        int m = r + 8 * lh;           // C/D layout: VGPR r <-> row m
        int b = rb + m;
        float f  = sigmoidf_(sQ[0][m] * vwpf + vbpf) + accF[r];
        float i_ = sigmoidf_(sQ[1][m] * vwpi + vbpi) + accI[r];
        float g_ = tanhf    (sQ[2][m] * vwpg + vbpg) + accG[r];
        float o_ = sigmoidf_(sQ[3][m] * vwpo + vbpo) + accO[r];
        float cp = (t > 0) ? cprev[(size_t)b * H_ + gn] : 0.0f;
        float cn = f * cp + i_ * g_;
        float hn = o_ * tanhf(cn);
        cnew[(size_t)b * H_ + gn]  = cn;
        ysrow[(size_t)b * H_ + gn] = hn;
        if (t == T_ - 1) {
            out_hx[(size_t)b * H_ + gn] = hn;
            out_cx[(size_t)b * H_ + gn] = cn;
        }
    }
}

// ---------------------------------------------------------------------------
extern "C" void kernel_launch(void* const* d_in, const int* in_sizes, int n_in,
                              void* d_out, int out_size, void* d_ws, size_t ws_size,
                              hipStream_t stream)
{
    const float* x    = (const float*)d_in[0];
    const float* Wf_l = (const float*)d_in[1];
    const float* bf_l = (const float*)d_in[2];
    const float* Wf_p = (const float*)d_in[3];
    const float* bf_p = (const float*)d_in[4];
    const float* Wf_r = (const float*)d_in[5];
    const float* Wi_l = (const float*)d_in[6];
    const float* bi_l = (const float*)d_in[7];
    const float* Wi_p = (const float*)d_in[8];
    const float* bi_p = (const float*)d_in[9];
    const float* Wi_r = (const float*)d_in[10];
    const float* Wg_l = (const float*)d_in[11];
    const float* bg_l = (const float*)d_in[12];
    const float* Wg_p = (const float*)d_in[13];
    const float* bg_p = (const float*)d_in[14];
    const float* Wg_r = (const float*)d_in[15];
    const float* Wo_l = (const float*)d_in[16];
    const float* bo_l = (const float*)d_in[17];
    const float* Wo_p = (const float*)d_in[18];
    const float* bo_p = (const float*)d_in[19];
    const float* Wo_r = (const float*)d_in[20];

    float* out    = (float*)d_out;
    float* ys     = out;                                   // (T,B,H)
    float* out_hx = out + (size_t)T_ * B_ * H_;            // (B,H)
    float* out_cx = out_hx + (size_t)B_ * H_;              // (B,H)

    // workspace layout: wcat (2 MiB bf16) | xdot (1 MiB f32) | c ping-pong (1 MiB)
    char* ws = (char*)d_ws;
    unsigned short* wcat = (unsigned short*)ws;
    float* xdot = (float*)(ws + (size_t)4 * H_ * H_ * sizeof(unsigned short));
    float* cbuf = (float*)(ws + (size_t)4 * H_ * H_ * sizeof(unsigned short)
                              + (size_t)T_ * B_ * 4 * sizeof(float));

    qlstm_prep_w<<<(4 * H_ * H_ + 255) / 256, 256, 0, stream>>>(
        Wf_r, Wi_r, Wg_r, Wo_r, wcat);
    qlstm_prep_xdot<<<(T_ * B_ + 255) / 256, 256, 0, stream>>>(
        x, Wf_l, Wi_l, Wg_l, Wo_l, bf_l, bi_l, bg_l, bo_l, xdot);

    dim3 grid(16, 4);
    for (int t = 0; t < T_; ++t) {
        float* cp = cbuf + (size_t)(t & 1) * B_ * H_;
        float* cn = cbuf + (size_t)((t + 1) & 1) * B_ * H_;
        qlstm_step<<<grid, 256, 0, stream>>>(
            t, wcat, xdot, Wf_l, Wi_l, Wg_l, Wo_l,
            Wf_p, bf_p, Wi_p, bi_p, Wg_p, bg_p, Wo_p, bo_p,
            cp, cn, ys, out_hx, out_cx);
    }
}
